// DeformableAlignmentBlock_30434138259570
// MI455X (gfx1250) — matrix operand order repaired and checked by reading, not compiled, not run on hardware
//
#include <hip/hip_runtime.h>
#include <hip/hip_bf16.h>
#include <stdint.h>

typedef __bf16 bf16;
typedef __attribute__((ext_vector_type(16))) bf16  v16bf;
typedef __attribute__((ext_vector_type(8)))  bf16  v8bf;
typedef __attribute__((ext_vector_type(8)))  float v8f;

typedef void __attribute__((address_space(3))) as3_void;
typedef void __attribute__((address_space(1))) as1_void;

typedef unsigned int u32x4 __attribute__((ext_vector_type(4)));
typedef int          i32x8 __attribute__((ext_vector_type(8)));
typedef int          i32x4 __attribute__((ext_vector_type(4)));

#define BATCH   4
#define CH      256
#define IMG_H   64
#define IMG_W   64
#define NPIX    (BATCH * IMG_H * IMG_W)   // 16384
#define HEADS   8
#define HD      32
#define NPTS    9
#define NGROUPS 4
#define CONV_K  (192 * 9)                 // 1728 per group
#define OFF_O   216                       // HEADS*NPTS*3
#define OFF_OP  256                       // padded to GEMM tile
#define GEMM_K  256

// ---------------------------------------------------------------------------
// TDM descriptor helper: 2-D bf16 tile (tile_w elements per row, tile_h rows,
// row stride = ld elements) from global -> LDS.  Per CDNA5 ISA.
// This toolchain exposes the 6-arg builtin (groups 0..3 + extra + cpol).
// ---------------------------------------------------------------------------
#if __has_builtin(__builtin_amdgcn_tensor_load_to_lds) && __has_builtin(__builtin_amdgcn_s_wait_tensorcnt)
#define HAVE_TDM 1
__device__ __forceinline__
void tdm_load_2d_bf16(const bf16* gsrc, unsigned int lds_addr,
                      unsigned int tile_w, unsigned int tile_h, unsigned int ld)
{
    unsigned long long ga = (unsigned long long)(uintptr_t)gsrc;
    u32x4 g0;
    g0[0] = 1u;                                   // count=1, user descriptor
    g0[1] = lds_addr;                             // lds_addr (bytes)
    g0[2] = (unsigned int)ga;                     // global_addr[31:0]
    g0[3] = (unsigned int)((ga >> 32) & 0x01FFFFFFu) | (2u << 30);  // [56:32] | type=2
    i32x8 g1;
    g1[0] = (int)(1u << 16);                      // workgroup_mask=0, data_size=1 (2 bytes)
    g1[1] = (int)(ld << 16);                      // tensor_dim0[15:0]
    g1[2] = (int)((ld >> 16) | (tile_h << 16));   // tensor_dim0[31:16] | tensor_dim1[15:0]
    g1[3] = (int)((tile_h >> 16) | (tile_w << 16)); // tensor_dim1[31:16] | tile_dim0
    g1[4] = (int)tile_h;                          // tile_dim1 | tile_dim2=0
    g1[5] = (int)ld;                              // tensor_dim0_stride[31:0]
    g1[6] = 0;                                    // stride[47:32] | dim1_stride lo
    g1[7] = 0;
    i32x4 zz; zz[0] = 0; zz[1] = 0; zz[2] = 0; zz[3] = 0;
    i32x8 z8; z8[0] = 0; z8[1] = 0; z8[2] = 0; z8[3] = 0;
               z8[4] = 0; z8[5] = 0; z8[6] = 0; z8[7] = 0;
    __builtin_amdgcn_tensor_load_to_lds(g0, g1, zz, zz, z8, 0);
}
#else
#define HAVE_TDM 0
#endif

#if __has_builtin(__builtin_amdgcn_global_load_async_to_lds_b128)
#define HAVE_ASYNC 1
#else
#define HAVE_ASYNC 0
#endif

// Stage a [tile_h][tile_w] bf16 panel (row stride ld) into LDS, cooperatively
// across nthr threads.  Chooses TDM -> async-copy -> sync fallback.
__device__ __forceinline__
void stage_panel(const bf16* gsrc, bf16* lds_dst, int tile_w, int tile_h,
                 int ld, int tid, int nthr, int issuer_wave)
{
#if HAVE_TDM
    if ((tid >> 5) == issuer_wave) {
        unsigned int lbase =
            (unsigned int)(uintptr_t)(as3_void*)lds_dst;
        tdm_load_2d_bf16(gsrc, lbase, (unsigned int)tile_w,
                         (unsigned int)tile_h, (unsigned int)ld);
        __builtin_amdgcn_s_wait_tensorcnt(0);
    }
#elif HAVE_ASYNC
    int chunks_per_row = tile_w / 8;
    int total = tile_h * chunks_per_row;
    for (int t = tid; t < total; t += nthr) {
        int row = t / chunks_per_row;
        int seg = t % chunks_per_row;
        __builtin_amdgcn_global_load_async_to_lds_b128(
            (as1_void*)(uintptr_t)(gsrc + (size_t)row * ld + seg * 8),
            (as3_void*)(lds_dst + (size_t)row * tile_w + seg * 8),
            0, 0);
    }
#if __has_builtin(__builtin_amdgcn_s_wait_asynccnt)
    __builtin_amdgcn_s_wait_asynccnt(0);
#else
    asm volatile("s_wait_asynccnt 0x0" ::: "memory");
#endif
#else
    int chunks_per_row = tile_w / 8;
    int total = tile_h * chunks_per_row;
    for (int t = tid; t < total; t += nthr) {
        int row = t / chunks_per_row;
        int seg = t % chunks_per_row;
        *(v8bf*)(lds_dst + (size_t)row * tile_w + seg * 8) =
            *(const v8bf*)(gsrc + (size_t)row * ld + seg * 8);
    }
#endif
}

// ---------------------------------------------------------------------------
// Pack NCHW f32 -> NHWC bf16 via LDS tile transpose.
// ---------------------------------------------------------------------------
__global__ __launch_bounds__(256)
void pack_feat_kernel(const float* __restrict__ pre, const float* __restrict__ post,
                      bf16* __restrict__ Xpre, bf16* __restrict__ Xpost,
                      bf16* __restrict__ Feat)
{
    __shared__ float tp[32][65];
    __shared__ float tq[32][65];
    int bi = blockIdx.x >> 9;
    int ct = (blockIdx.x >> 6) & 7;
    int pt = blockIdx.x & 63;
    int tid = threadIdx.x;

    #pragma unroll
    for (int r = 0; r < 8; ++r) {
        int cl = r * 4 + (tid >> 6);
        int pl = tid & 63;
        size_t src = ((size_t)(bi * CH + ct * 32 + cl) << 12) + pt * 64 + pl;
        tp[cl][pl] = pre[src];
        tq[cl][pl] = post[src];
    }
    __syncthreads();
    #pragma unroll
    for (int r = 0; r < 8; ++r) {
        int pl = r * 8 + (tid >> 5);
        int cl = tid & 31;
        float pv = tp[cl][pl];
        float qv = tq[cl][pl];
        int n = bi * 4096 + pt * 64 + pl;
        int c = ct * 32 + cl;
        Xpre [(size_t)n * CH  + c]       = (bf16)pv;
        Xpost[(size_t)n * CH  + c]       = (bf16)qv;
        Feat [(size_t)n * 768 + c]       = (bf16)pv;
        Feat [(size_t)n * 768 + 256 + c] = (bf16)qv;
        Feat [(size_t)n * 768 + 512 + c] = (bf16)(qv - pv);
    }
}

// ---------------------------------------------------------------------------
// Pack / transpose weights to bf16 [K][O] layouts.
// ---------------------------------------------------------------------------
__global__ __launch_bounds__(256)
void pack_weights_kernel(const float* __restrict__ wq, const float* __restrict__ wk,
                         const float* __restrict__ wv, const float* __restrict__ woff1,
                         const float* __restrict__ woff2, const float* __restrict__ wout,
                         bf16* __restrict__ Wq_t, bf16* __restrict__ Wkv_t,
                         bf16* __restrict__ Wct, bf16* __restrict__ Woff2_t,
                         bf16* __restrict__ Wout_t)
{
    int idx = blockIdx.x * 256 + threadIdx.x;
    const int R0 = 256 * 256;
    const int R1 = 256 * 512;
    const int R2 = NGROUPS * CONV_K * 64;
    const int R3 = 256 * OFF_OP;
    const int R4 = 256 * 256;

    if (idx < R0) {
        int c = idx >> 8, o = idx & 255;
        Wq_t[c * 256 + o] = (bf16)wq[o * 256 + c];
        return;
    }
    idx -= R0;
    if (idx < R1) {
        int c = idx >> 9, o = idx & 511;
        float v = (o < 256) ? wk[o * 256 + c] : wv[(o - 256) * 256 + c];
        Wkv_t[c * 512 + o] = (bf16)v;
        return;
    }
    idx -= R1;
    if (idx < R2) {
        int oc = idx & 63;
        int t  = idx >> 6;
        int kk = t % CONV_K;
        int g  = t / CONV_K;
        int cell = kk / 192;
        int c    = kk % 192;
        Wct[(size_t)idx] = (bf16)woff1[(((size_t)(g * 64 + oc) * 192 + c) * 9) + cell];
        return;
    }
    idx -= R2;
    if (idx < R3) {
        int c = idx >> 8, o = idx & 255;
        Woff2_t[c * OFF_OP + o] = (bf16)((o < OFF_O) ? woff2[o * 256 + c] : 0.0f);
        return;
    }
    idx -= R3;
    if (idx < R4) {
        int c = idx >> 8, o = idx & 255;
        Wout_t[c * 256 + o] = (bf16)wout[o * 256 + c];
    }
}

// ---------------------------------------------------------------------------
// bf16 WMMA GEMM, K=256.  Weight panel [256][64] staged once into LDS via TDM
// (fallback: async-copy / sync), shared by all 4 waves.  A from global.
// Epilogue modes: 0 plain f32, 1 kv-remap, 2 +bias, 3 BN2+residual NCHW.
// ---------------------------------------------------------------------------
__global__ __launch_bounds__(128)
void gemm_wmma_kernel(const bf16* __restrict__ X, const bf16* __restrict__ Wt,
                      int ldx, int ldw, int ldo,
                      float* __restrict__ Y, int mode,
                      const float* __restrict__ bias,
                      const float* __restrict__ gamma, const float* __restrict__ beta,
                      const float* __restrict__ mean,  const float* __restrict__ var,
                      const float* __restrict__ resid)
{
    __shared__ bf16 Bsm[GEMM_K * 64];        // 32 KB weight panel

    int lane  = threadIdx.x & 31;
    int wave  = threadIdx.x >> 5;
    int ntile = blockIdx.x * 16;
    int obase = blockIdx.y * 64;
    int otile = obase + wave * 16;
    int m     = lane & 15;
    int kboff = (lane >> 4) << 3;

    stage_panel(Wt + obase, Bsm, 64, GEMM_K, ldw, threadIdx.x, 128, 0);
    __syncthreads();

    const bf16* arow = X   + (size_t)(ntile + m) * ldx;
    const bf16* bsm  = Bsm + kboff * 64 + wave * 16 + m;    // rows k+kboff.., wave's col m

    v8f acc;
    #pragma unroll
    for (int i = 0; i < 8; ++i) acc[i] = 0.0f;

    #pragma unroll
    for (int k = 0; k < GEMM_K; k += 32) {
        v8bf a0 = *(const v8bf*)(arow + k + kboff);
        v8bf a1 = *(const v8bf*)(arow + k + kboff + 16);
        v8bf b0, b1;
        #pragma unroll
        for (int i = 0; i < 8; ++i) { b0[i] = bsm[(k + i) * 64]; b1[i] = bsm[(k + 16 + i) * 64]; }
        if (k + 32 < GEMM_K)
            __builtin_prefetch(arow + k + 32 + kboff, 0, 1);
        v16bf a, b;
        #pragma unroll
        for (int i = 0; i < 8; ++i) {
            a[i] = a0[i]; a[i + 8] = a1[i];
            b[i] = b0[i]; b[i + 8] = b1[i];
        }
        acc = __builtin_amdgcn_wmma_f32_16x16x32_bf16(false, a, false, b,
                                                      (short)0, acc, false, false);
    }

    int ocol = otile + m;
    int mb   = (lane >> 4) << 3;

    if (mode == 0) {
        #pragma unroll
        for (int i = 0; i < 8; ++i)
            Y[(size_t)(ntile + i + mb) * ldo + ocol] = acc[i];
    } else if (mode == 1) {
        int dsto = ((ocol & 255) >> 5) * 64 + ((ocol >> 8) << 5) + (ocol & 31);
        #pragma unroll
        for (int i = 0; i < 8; ++i)
            Y[(size_t)(ntile + i + mb) * 512 + dsto] = acc[i];
    } else if (mode == 2) {
        float bv = (ocol < OFF_O) ? bias[ocol] : 0.0f;
        #pragma unroll
        for (int i = 0; i < 8; ++i)
            Y[(size_t)(ntile + i + mb) * ldo + ocol] = acc[i] + bv;
    } else {
        float iv = gamma[ocol] * rsqrtf(var[ocol] + 1e-5f);
        float sh = beta[ocol] - mean[ocol] * iv;
        #pragma unroll
        for (int i = 0; i < 8; ++i) {
            int row = ntile + i + mb;
            int bb  = row >> 12;
            int rem = row & 4095;
            size_t di = ((size_t)(bb * CH + ocol) << 12) + rem;
            Y[di] = acc[i] * iv + sh + resid[di];
        }
    }
}

// ---------------------------------------------------------------------------
// Grouped 3x3 conv as implicit WMMA GEMM (K=1728/group), SAME padding.
// Weight panel [1728][16] staged into LDS via TDM, BN1+SiLU epilogue.
// ---------------------------------------------------------------------------
__global__ __launch_bounds__(32)
void conv_wmma_kernel(const bf16* __restrict__ Feat, const bf16* __restrict__ Wct,
                      const float* __restrict__ gam, const float* __restrict__ bet,
                      const float* __restrict__ mu,  const float* __restrict__ var,
                      bf16* __restrict__ Hout)
{
    __shared__ bf16 Bsm[CONV_K * 16];        // 54 KB weight panel

    int lane  = threadIdx.x & 31;
    int ntile = blockIdx.x * 16;
    int ot    = blockIdx.y;
    int g     = ot >> 2;
    int osub  = (ot & 3) * 16;
    int m     = lane & 15;
    int kboff = (lane >> 4) << 3;

    const bf16* wpanel = Wct + (size_t)g * CONV_K * 64 + osub;
    stage_panel(wpanel, Bsm, 16, CONV_K, 64, threadIdx.x, 32, 0);
    __syncthreads();

    int pix = ntile + m;
    int bb  = pix >> 12;
    int rem = pix & 4095;
    int py  = rem >> 6;
    int px  = rem & 63;

    v8f acc;
    #pragma unroll
    for (int i = 0; i < 8; ++i) acc[i] = 0.0f;

    for (int kc = 0; kc < 54; ++kc) {
        int cell = kc / 6;
        int ky   = cell / 3 - 1;
        int kx   = cell % 3 - 1;
        int coff = (kc % 6) * 32;
        int sy = py + ky, sx = px + kx;
        bool ok = ((unsigned)sy < (unsigned)IMG_H) && ((unsigned)sx < (unsigned)IMG_W);

        v8bf a0, a1;
        if (ok) {
            const bf16* p = Feat + ((size_t)((bb * IMG_H + sy) * IMG_W + sx)) * 768
                          + g * 192 + coff + kboff;
            a0 = *(const v8bf*)p;
            a1 = *(const v8bf*)(p + 16);
        } else {
            #pragma unroll
            for (int i = 0; i < 8; ++i) { a0[i] = (bf16)0.0f; a1[i] = (bf16)0.0f; }
        }
        const bf16* bp = Bsm + (size_t)(cell * 192 + coff + kboff) * 16 + m;
        v8bf b0, b1;
        #pragma unroll
        for (int i = 0; i < 8; ++i) { b0[i] = bp[i * 16]; b1[i] = bp[(16 + i) * 16]; }

        v16bf a, b;
        #pragma unroll
        for (int i = 0; i < 8; ++i) {
            a[i] = a0[i]; a[i + 8] = a1[i];
            b[i] = b0[i]; b[i + 8] = b1[i];
        }
        acc = __builtin_amdgcn_wmma_f32_16x16x32_bf16(false, a, false, b,
                                                      (short)0, acc, false, false);
    }

    int oc = g * 64 + osub + m;
    float iv = gam[oc] * rsqrtf(var[oc] + 1e-5f);
    float sh = bet[oc] - mu[oc] * iv;
    int mb = (lane >> 4) << 3;
    #pragma unroll
    for (int i = 0; i < 8; ++i) {
        float val = acc[i] * iv + sh;
        float s   = val / (1.0f + __expf(-val));     // SiLU
        Hout[(size_t)(ntile + i + mb) * CH + oc] = (bf16)s;
    }
}

// ---------------------------------------------------------------------------
// Deformable sampling + attention (one wave per (pixel, head), lane = channel).
// ---------------------------------------------------------------------------
__global__ __launch_bounds__(256)
void attn_sample_kernel(const float* __restrict__ q, const float* __restrict__ kv,
                        const float* __restrict__ logits, bf16* __restrict__ aligned)
{
    int wave = threadIdx.x >> 5;
    int lane = threadIdx.x & 31;
    int task = blockIdx.x * 8 + wave;
    int head = task & 7;
    int pix  = task >> 3;
    int bb   = pix >> 12;
    int rem  = pix & 4095;
    int py   = rem >> 6;
    int px   = rem & 63;

    float qc = q[(size_t)pix * CH + head * HD + lane];
    const float* lg   = logits + (size_t)pix * OFF_OP + head * (NPTS * 3);
    const float* rowb = kv + ((size_t)bb << 21);
    int co = head * 64 + lane;

    float scores[NPTS];
    float vvals[NPTS];

    #pragma unroll
    for (int p = 0; p < NPTS; ++p) {
        float ox = tanhf(lg[p * 3 + 0]);
        float oy = tanhf(lg[p * 3 + 1]);
        float bias = lg[p * 3 + 2];
        float dx = (float)((p % 3) * 2 - 2);
        float dy = (float)((p / 3) * 2 - 2);
        float sx = (float)px + dx + ox;
        float sy = (float)py + dy + oy;

        float x0f = floorf(sx), y0f = floorf(sy);
        float wx = sx - x0f, wy = sy - y0f;
        int x0 = min(max((int)x0f, 0), IMG_W - 1);
        int x1 = min(max((int)x0f + 1, 0), IMG_W - 1);
        int y0 = min(max((int)y0f, 0), IMG_H - 1);
        int y1 = min(max((int)y0f + 1, 0), IMG_H - 1);

        int i00 = (y0 * IMG_W + x0) * 512 + co;
        int i01 = (y0 * IMG_W + x1) * 512 + co;
        int i10 = (y1 * IMG_W + x0) * 512 + co;
        int i11 = (y1 * IMG_W + x1) * 512 + co;

        float k00 = rowb[i00],      k01 = rowb[i01];
        float k10 = rowb[i10],      k11 = rowb[i11];
        float v00 = rowb[i00 + 32], v01 = rowb[i01 + 32];
        float v10 = rowb[i10 + 32], v11 = rowb[i11 + 32];

        float ks = (k00 * (1.f - wx) + k01 * wx) * (1.f - wy)
                 + (k10 * (1.f - wx) + k11 * wx) * wy;
        float vs = (v00 * (1.f - wx) + v01 * wx) * (1.f - wy)
                 + (v10 * (1.f - wx) + v11 * wx) * wy;

        float partial = qc * ks;
        #pragma unroll
        for (int off = 16; off > 0; off >>= 1)
            partial += __shfl_xor(partial, off, 32);

        scores[p] = partial * 0.17677669529663687f + bias;
        vvals[p]  = vs;
    }

    float mx = scores[0];
    #pragma unroll
    for (int p = 1; p < NPTS; ++p) mx = fmaxf(mx, scores[p]);
    float e[NPTS];
    float s = 0.0f;
    #pragma unroll
    for (int p = 0; p < NPTS; ++p) { e[p] = __expf(scores[p] - mx); s += e[p]; }
    float inv = 1.0f / s;
    float accv = 0.0f;
    #pragma unroll
    for (int p = 0; p < NPTS; ++p) accv += e[p] * inv * vvals[p];

    aligned[(size_t)pix * CH + head * HD + lane] = (bf16)accv;
}

// ---------------------------------------------------------------------------
extern "C" void kernel_launch(void* const* d_in, const int* in_sizes, int n_in,
                              void* d_out, int out_size, void* d_ws, size_t ws_size,
                              hipStream_t stream)
{
    const float* pre   = (const float*)d_in[0];
    const float* post  = (const float*)d_in[1];
    const float* wq    = (const float*)d_in[2];
    const float* wk    = (const float*)d_in[3];
    const float* wv    = (const float*)d_in[4];
    const float* woff1 = (const float*)d_in[5];
    const float* g1    = (const float*)d_in[6];
    const float* be1   = (const float*)d_in[7];
    const float* m1    = (const float*)d_in[8];
    const float* v1    = (const float*)d_in[9];
    const float* woff2 = (const float*)d_in[10];
    const float* boff2 = (const float*)d_in[11];
    const float* wout  = (const float*)d_in[12];
    const float* g2    = (const float*)d_in[13];
    const float* be2   = (const float*)d_in[14];
    const float* m2    = (const float*)d_in[15];
    const float* v2    = (const float*)d_in[16];
    float* out = (float*)d_out;

    char* ws = (char*)d_ws;
    size_t off = 0;
    auto alloc = [&](size_t bytes) -> void* {
        void* p = ws + off;
        off += (bytes + 255) & ~(size_t)255;
        return p;
    };

    bf16*  Xpre    = (bf16*) alloc((size_t)NPIX * CH * 2);
    bf16*  Xpost   = (bf16*) alloc((size_t)NPIX * CH * 2);
    bf16*  Feat    = (bf16*) alloc((size_t)NPIX * 768 * 2);
    bf16*  Hbf     = (bf16*) alloc((size_t)NPIX * CH * 2);
    bf16*  Albf    = (bf16*) alloc((size_t)NPIX * CH * 2);
    bf16*  Wq_t    = (bf16*) alloc(256 * 256 * 2);
    bf16*  Wkv_t   = (bf16*) alloc(256 * 512 * 2);
    bf16*  Wct     = (bf16*) alloc((size_t)NGROUPS * CONV_K * 64 * 2);
    bf16*  Woff2_t = (bf16*) alloc(256 * OFF_OP * 2);
    bf16*  Wout_t  = (bf16*) alloc(256 * 256 * 2);
    float* Qf      = (float*)alloc((size_t)NPIX * CH * 4);
    float* KVf     = (float*)alloc((size_t)NPIX * 512 * 4);
    float* Lg      = (float*)alloc((size_t)NPIX * OFF_OP * 4);

    pack_feat_kernel<<<dim3(BATCH * 8 * 64), dim3(256), 0, stream>>>(
        pre, post, Xpre, Xpost, Feat);

    pack_weights_kernel<<<dim3(3008), dim3(256), 0, stream>>>(
        wq, wk, wv, woff1, woff2, wout, Wq_t, Wkv_t, Wct, Woff2_t, Wout_t);

    gemm_wmma_kernel<<<dim3(NPIX / 16, 4), dim3(128), 0, stream>>>(
        Xpost, Wq_t, 256, 256, 256, Qf, 0,
        nullptr, nullptr, nullptr, nullptr, nullptr, nullptr);

    gemm_wmma_kernel<<<dim3(NPIX / 16, 8), dim3(128), 0, stream>>>(
        Xpre, Wkv_t, 256, 512, 512, KVf, 1,
        nullptr, nullptr, nullptr, nullptr, nullptr, nullptr);

    conv_wmma_kernel<<<dim3(NPIX / 16, 16), dim3(32), 0, stream>>>(
        Feat, Wct, g1, be1, m1, v1, Hbf);

    gemm_wmma_kernel<<<dim3(NPIX / 16, 4), dim3(128), 0, stream>>>(
        Hbf, Woff2_t, 256, 256, OFF_OP, Lg, 2,
        boff2, nullptr, nullptr, nullptr, nullptr, nullptr);

    attn_sample_kernel<<<dim3((NPIX * HEADS) / 8), dim3(256), 0, stream>>>(
        Qf, KVf, Lg, Albf);

    gemm_wmma_kernel<<<dim3(NPIX / 16, 4), dim3(128), 0, stream>>>(
        Albf, Wout_t, 256, 256, 256, out, 3,
        nullptr, g2, be2, m2, v2, pre);
}
